// MCDMultivariateNormal_10067403341934
// MI455X (gfx1250) — matrix operand order repaired and checked by reading, not compiled, not run on hardware
//
#include <hip/hip_runtime.h>
#include <hip/hip_bf16.h>
#include <stdint.h>

// Problem constants (from reference): B=8192, HID=1024, FH=96, O=8, E=36
#define NB   8192
#define HID  1024
#define FH   96
#define O_   8
#define E_   36

typedef __bf16 v16bf __attribute__((ext_vector_type(16)));
typedef __bf16 v8bf  __attribute__((ext_vector_type(8)));
typedef float  v8f   __attribute__((ext_vector_type(8)));

// float -> bf16 (round-to-nearest-even), returned as raw u16 so host pass never
// needs arithmetic __bf16.
__device__ __forceinline__ uint16_t f32_to_bf16_bits(float f) {
  union { float f; uint32_t u; } v; v.f = f;
  uint32_t r = v.u + 0x7FFFu + ((v.u >> 16) & 1u);
  return (uint16_t)(r >> 16);
}

// ---------------------------------------------------------------------------
// Elementwise fp32 -> bf16 (for activations z, row-major kept)
// ---------------------------------------------------------------------------
__global__ __launch_bounds__(256) void cvt_bf16_kernel(
    const float* __restrict__ in, uint16_t* __restrict__ out, int n) {
  int i = blockIdx.x * blockDim.x + threadIdx.x;
  if (i < n) out[i] = f32_to_bf16_bits(in[i]);
}

// ---------------------------------------------------------------------------
// Weight transpose + convert: in is K x N row-major fp32, out is N x K bf16
// (i.e. column-major B). LDS-tiled to keep both sides coalesced.
// ---------------------------------------------------------------------------
__global__ __launch_bounds__(256) void transpose_cvt_bf16_kernel(
    const float* __restrict__ in, uint16_t* __restrict__ out, int K, int N) {
  __shared__ float tile[32][33];
  const int n0 = blockIdx.x * 32;
  const int k0 = blockIdx.y * 32;
  const int tx = threadIdx.x;  // 0..31
  const int ty = threadIdx.y;  // 0..7
#pragma unroll
  for (int r = 0; r < 32; r += 8)
    tile[ty + r][tx] = in[(size_t)(k0 + ty + r) * N + (n0 + tx)];
  __syncthreads();
#pragma unroll
  for (int r = 0; r < 32; r += 8)
    out[(size_t)(n0 + ty + r) * K + (k0 + tx)] =
        f32_to_bf16_bits(tile[tx][ty + r]);
}

// ---------------------------------------------------------------------------
// bf16 WMMA GEMM:  Out(MxN) = act( A(MxK,row-major) * B + bias )
// B supplied transposed: Bt is N x K bf16 (so each lane's B fragment is a
// contiguous 16-element K run, in the same k order as A).
// 4 waves / block; wave w owns a 16x64 tile (4x v_wmma_f32_16x16x32_bf16 per
// 32-wide K step). MODE: 0 = +bias,SiLU -> bf16 ; 1 = +bias -> bf16 ;
// 2 = +bias -> fp32.
// M % 64 == 0, N % 64 == 0, K % 32 == 0 (holds for all four GEMMs here).
// ---------------------------------------------------------------------------
template <int MODE>
__global__ __launch_bounds__(128) void gemm_wmma_bf16_kernel(
    const uint16_t* __restrict__ Au, const uint16_t* __restrict__ Btu,
    const float* __restrict__ bias, void* __restrict__ Out,
    int M, int N, int K) {
  const __bf16* __restrict__ A  = (const __bf16*)Au;
  const __bf16* __restrict__ Bt = (const __bf16*)Btu;

  const int lane = threadIdx.x & 31;
  const int wave = threadIdx.x >> 5;
  const int half = lane >> 4;   // 0: lanes 0-15, 1: lanes 16-31
  const int l16  = lane & 15;
  const int m0 = blockIdx.y * 64 + wave * 16;
  const int n0 = blockIdx.x * 64;

  v8f acc[4] = {v8f{}, v8f{}, v8f{}, v8f{}};

  // A: lane holds row (m0+l16); lo chunk K=[kk+half*16, +8), hi chunk
  // K=[kk+half*16+8, +8)  — contiguous K run of 16, matching B.
  const __bf16* aRow = A + (size_t)(m0 + l16) * K + half * 16;
  // B: lane holds column (n0+16t+l16), contiguous K run of 16 starting at
  // kk + half*16 — same k order as A.
  const __bf16* bp[4];
#pragma unroll
  for (int t = 0; t < 4; ++t)
    bp[t] = Bt + (size_t)(n0 + 16 * t + l16) * K + half * 16;

  for (int kk = 0; kk < K; kk += 32) {
    v8bf alo = *(const v8bf*)(aRow + kk);
    v8bf ahi = *(const v8bf*)(aRow + kk + 8);
    v16bf a = __builtin_shufflevector(alo, ahi, 0, 1, 2, 3, 4, 5, 6, 7, 8, 9,
                                      10, 11, 12, 13, 14, 15);
    // speculative prefetch of the K block two steps ahead (global_prefetch_b8)
    __builtin_prefetch(aRow + kk + 64, 0, 0);
    __builtin_prefetch(bp[0] + kk + 64, 0, 0);
#pragma unroll
    for (int t = 0; t < 4; ++t) {
      v16bf b = *(const v16bf*)(bp[t] + kk);
      acc[t] = __builtin_amdgcn_wmma_f32_16x16x32_bf16(
          false, a, false, b, (short)0, acc[t], false, false);
    }
  }

  // Epilogue. C/D layout: VGPR r, lanes 0-15 -> (M=r, N=lane),
  // lanes 16-31 -> (M=8+r, N=lane-16).
#pragma unroll
  for (int t = 0; t < 4; ++t) {
    const int col = n0 + 16 * t + l16;
    const float bv = bias[col];
#pragma unroll
    for (int r = 0; r < 8; ++r) {
      const int row = m0 + half * 8 + r;
      float v = acc[t][r] + bv;
      if (MODE == 0) {
        v = v / (1.0f + __expf(-v));  // SiLU
        ((uint16_t*)Out)[(size_t)row * N + col] = f32_to_bf16_bits(v);
      } else if (MODE == 1) {
        ((uint16_t*)Out)[(size_t)row * N + col] = f32_to_bf16_bits(v);
      } else {
        ((float*)Out)[(size_t)row * N + col] = v;
      }
    }
  }
}

// ---------------------------------------------------------------------------
// Per-(b,fh) head: d = exp(0.5*eta_d), D = diag(d), T = unit upper-tri with
// eta_T, precision = T^T diag(1/d) T. One thread per (b,fh); O=8 fully
// unrolled. Memory-bound on the 3x64-float stores.
// ---------------------------------------------------------------------------
__global__ __launch_bounds__(256) void mvn_finalize_kernel(
    const float* __restrict__ logvar,  // (B*FH) x E
    float* __restrict__ outP, float* __restrict__ outD,
    float* __restrict__ outT) {
  const int t = blockIdx.x * blockDim.x + threadIdx.x;
  if (t >= NB * FH) return;
  const float* lv = logvar + (size_t)t * E_;

  float d[O_], invd[O_];
#pragma unroll
  for (int i = 0; i < O_; ++i) {
    const float e = lv[i];
    d[i] = __expf(0.5f * e);
    invd[i] = __expf(-0.5f * e);
  }

  float T[O_][O_];
#pragma unroll
  for (int i = 0; i < O_; ++i)
#pragma unroll
    for (int j = 0; j < O_; ++j) T[i][j] = (i == j) ? 1.0f : 0.0f;
  {
    int idx = O_;  // eta_T starts after the O diagonal entries
#pragma unroll
    for (int i = 0; i < O_; ++i)
#pragma unroll
      for (int j = i + 1; j < O_; ++j) T[i][j] = lv[idx++];
  }

  float* pT = outT + (size_t)t * 64;
  float* pD = outD + (size_t)t * 64;
  float* pP = outP + (size_t)t * 64;
#pragma unroll
  for (int i = 0; i < O_; ++i)
#pragma unroll
    for (int j = 0; j < O_; ++j) {
      pT[i * 8 + j] = T[i][j];
      pD[i * 8 + j] = (i == j) ? d[i] : 0.0f;
    }

  // P[i][k] = sum_{j<=min(i,k)} T[j][i] * T[j][k] / d[j]
#pragma unroll
  for (int i = 0; i < O_; ++i)
#pragma unroll
    for (int k = 0; k < O_; ++k) {
      const int jmax = i < k ? i : k;
      float s = 0.0f;
#pragma unroll
      for (int j = 0; j <= jmax; ++j) s += T[j][i] * T[j][k] * invd[j];
      pP[i * 8 + k] = s;
    }
}

// ---------------------------------------------------------------------------
extern "C" void kernel_launch(void* const* d_in, const int* in_sizes, int n_in,
                              void* d_out, int out_size, void* d_ws,
                              size_t ws_size, hipStream_t stream) {
  const float* z  = (const float*)d_in[0];
  const float* W1 = (const float*)d_in[1];
  const float* b1 = (const float*)d_in[2];
  const float* W2 = (const float*)d_in[3];
  const float* b2 = (const float*)d_in[4];
  const float* Wm = (const float*)d_in[5];
  const float* bm = (const float*)d_in[6];
  const float* Wv = (const float*)d_in[7];
  const float* bv = (const float*)d_in[8];

  // Workspace layout (bytes). z/h1 region is dead by GEMM4 and is overlapped
  // by the fp32 log_var buffer. Total footprint ~143 MB.
  char* ws = (char*)d_ws;
  uint16_t* hb  = (uint16_t*)(ws + 0);          // h   : 8192x1024 bf16 (16 MB)
  uint16_t* W1t = (uint16_t*)(ws + 16777216);   // 1024x1024 bf16
  uint16_t* W2t = (uint16_t*)(ws + 16777216 + 2097152);
  uint16_t* Wmt = (uint16_t*)(ws + 16777216 + 4194304);          // 768x1024
  uint16_t* Wvt = (uint16_t*)(ws + 16777216 + 4194304 + 1572864); // 3456x1024
  uint16_t* zb  = (uint16_t*)(ws + 29622272);                // z  bf16 (16 MB)
  uint16_t* h1b = (uint16_t*)(ws + 29622272 + 16777216);     // h1 bf16 (16 MB)
  float* lvar   = (float*)(ws + 29622272);  // log_var fp32 113 MB (reuses z/h1)

  // Output regions: [means | precision | D | T], all fp32.
  float* outMeans = (float*)d_out;                 // 8192*96*8   = 6291456
  float* outP     = outMeans + 6291456;            // 8192*96*64  = 50331648
  float* outD     = outP + 50331648;
  float* outT     = outD + 50331648;

  const dim3 tb(32, 8);

  // Stage 0: precision conversion + weight transposes
  cvt_bf16_kernel<<<(NB * HID + 255) / 256, 256, 0, stream>>>(z, zb, NB * HID);
  transpose_cvt_bf16_kernel<<<dim3(1024 / 32, 1024 / 32), tb, 0, stream>>>(
      W1, W1t, 1024, 1024);
  transpose_cvt_bf16_kernel<<<dim3(1024 / 32, 1024 / 32), tb, 0, stream>>>(
      W2, W2t, 1024, 1024);
  transpose_cvt_bf16_kernel<<<dim3(768 / 32, 1024 / 32), tb, 0, stream>>>(
      Wm, Wmt, 1024, 768);
  transpose_cvt_bf16_kernel<<<dim3(3456 / 32, 1024 / 32), tb, 0, stream>>>(
      Wv, Wvt, 1024, 3456);

  // Stage 1: h1 = silu(z@W1 + b1)  (bf16)
  gemm_wmma_bf16_kernel<0><<<dim3(1024 / 64, NB / 64), 128, 0, stream>>>(
      zb, W1t, b1, h1b, NB, 1024, 1024);
  // Stage 2: h = h1@W2 + b2  (bf16)
  gemm_wmma_bf16_kernel<1><<<dim3(1024 / 64, NB / 64), 128, 0, stream>>>(
      h1b, W2t, b2, hb, NB, 1024, 1024);
  // Stage 3: means = h@Wm + bm  -> d_out (fp32, row-major == (B,FH,O))
  gemm_wmma_bf16_kernel<2><<<dim3(768 / 64, NB / 64), 128, 0, stream>>>(
      hb, Wmt, bm, outMeans, NB, 768, 1024);
  // Stage 4: log_var = h@Wv + bv -> ws (fp32)
  gemm_wmma_bf16_kernel<2><<<dim3(3456 / 64, NB / 64), 128, 0, stream>>>(
      hb, Wvt, bv, lvar, NB, 3456, 1024);

  // Stage 5: per-(b,fh) D / T / precision
  mvn_finalize_kernel<<<(NB * FH) / 256, 256, 0, stream>>>(lvar, outP, outD,
                                                           outT);
}
